// Critic_89575837925789
// MI455X (gfx1250) — matrix operand order repaired and checked by reading, not compiled, not run on hardware
//
#include <hip/hip_runtime.h>
#include <hip/hip_bf16.h>

typedef __attribute__((ext_vector_type(16))) __bf16 v16bf;
typedef __attribute__((ext_vector_type(8)))  float v8f;

#define B_   512
#define T_   512
#define C_   32
#define H_   128
#define G_   512      // 4H
#define KTOT 192      // 128 (h) + 32 (x) + 1 (std) + 31 (zero pad)

union U16 { unsigned int i[8]; v16bf v; };

__device__ __forceinline__ unsigned int f2bfu(float f) {
    unsigned int u = __float_as_uint(f);
    return (u + 0x7FFFu + ((u >> 16) & 1u)) >> 16;   // RNE
}
__device__ __forceinline__ float bf2f(unsigned short s) {
    return __uint_as_float(((unsigned int)s) << 16);
}

// ---------------- Kernel A: minibatch std feature: std_t[t] ----------------
__global__ __launch_bounds__(256) void std_kernel(const float* __restrict__ x,
                                                  float* __restrict__ std_t) {
    __shared__ float sS[256], sQ[256], stdc[32];
    const int t = blockIdx.x, tid = threadIdx.x;
    const int c = tid & 31, grp = tid >> 5;
    float s = 0.f, q = 0.f;
    for (int b = grp; b < B_; b += 8) {
        float v = x[((size_t)b * T_ + t) * C_ + c];
        s += v; q += v * v;
    }
    sS[tid] = s; sQ[tid] = q;
    __syncthreads();
    if (tid < 32) {
        float S = 0.f, Q = 0.f;
        for (int g = 0; g < 8; ++g) { S += sS[g * 32 + tid]; Q += sQ[g * 32 + tid]; }
        float mean = S * (1.f / B_);
        float var  = (Q - (float)B_ * mean * mean) * (1.f / (B_ - 1));
        stdc[tid]  = sqrtf(fmaxf(var, 0.f));
    }
    __syncthreads();
    if (tid == 0) {
        float a = 0.f;
        for (int i = 0; i < 32; ++i) a += stdc[i];
        std_t[t] = a * (1.f / 32.f);
    }
}

// ---------------- Kernel B: spectral-norm sigmas (torch-style 1 iter) ------
__global__ __launch_bounds__(256) void sn_kernel(
    const float* __restrict__ w_ih, const float* __restrict__ u_ih,
    const float* __restrict__ w_hh, const float* __restrict__ u_hh,
    const float* __restrict__ fc_w, const float* __restrict__ u_fc,
    float* __restrict__ sig) {
    __shared__ float tvec[128], vvec[128], red[256];
    __shared__ float bc;
    const int id = blockIdx.x, tid = threadIdx.x;
    const float* W; const float* u; int R, Cn;
    if (id == 0)      { W = w_ih; u = u_ih; R = G_; Cn = C_ + 1; }
    else if (id == 1) { W = w_hh; u = u_hh; R = G_; Cn = H_; }
    else              { W = fc_w; u = u_fc; R = 1;  Cn = H_; }
    // t = W^T u
    for (int c = tid; c < Cn; c += 256) {
        float a = 0.f;
        for (int r = 0; r < R; ++r) a += W[(size_t)r * Cn + c] * u[r];
        tvec[c] = a;
    }
    __syncthreads();
    red[tid] = (tid < Cn) ? tvec[tid] * tvec[tid] : 0.f;
    __syncthreads();
    for (int s = 128; s > 0; s >>= 1) { if (tid < s) red[tid] += red[tid + s]; __syncthreads(); }
    if (tid == 0) bc = 1.f / (sqrtf(red[0]) + 1e-12f);
    __syncthreads();
    const float invn = bc;
    for (int c = tid; c < Cn; c += 256) vvec[c] = tvec[c] * invn;
    __syncthreads();
    // sigma = ||Wv||^2 / (||Wv|| + eps)
    float acc = 0.f;
    for (int r = tid; r < R; r += 256) {
        float sd = 0.f;
        for (int c = 0; c < Cn; ++c) sd += W[(size_t)r * Cn + c] * vvec[c];
        acc += sd * sd;
    }
    red[tid] = acc;
    __syncthreads();
    for (int s = 128; s > 0; s >>= 1) { if (tid < s) red[tid] += red[tid + s]; __syncthreads(); }
    if (tid == 0) {
        float n2 = red[0];
        float sigma = n2 / (sqrtf(n2) + 1e-12f);
        sig[id] = 1.f / sigma;
    }
}

// ---------------- Kernel C: pack combined bf16 weights + bias + fc ---------
// wcomb layout: [n=0..511][k=0..191], k<128: Whh^T/sig_hh ; 128..159: Wih^T chans ;
// 160: Wih^T std-chan ; 161..191: zero.
__global__ __launch_bounds__(256) void pack_kernel(
    const float* __restrict__ w_ih, const float* __restrict__ w_hh,
    const float* __restrict__ b_ih, const float* __restrict__ b_hh,
    const float* __restrict__ fc_w, const float* __restrict__ sig,
    unsigned short* __restrict__ wcomb, float* __restrict__ biasg,
    float* __restrict__ fcw) {
    const int idx = blockIdx.x * 256 + threadIdx.x;  // 0 .. 512*192-1
    const int n = idx / KTOT, k = idx % KTOT;
    const float inv_ih = sig[0], inv_hh = sig[1];
    float v;
    if (k < H_)            v = w_hh[(size_t)n * H_ + k] * inv_hh;
    else if (k < H_ + C_)  v = w_ih[(size_t)n * (C_ + 1) + (k - H_)] * inv_ih;
    else if (k == H_ + C_) v = w_ih[(size_t)n * (C_ + 1) + C_] * inv_ih;
    else                   v = 0.f;
    wcomb[idx] = (unsigned short)f2bfu(v);
    if (idx < G_) biasg[idx] = b_ih[idx] + b_hh[idx];
    if (idx < H_) fcw[idx] = fc_w[idx] * sig[2];
}

// ---------------- Kernel D: persistent LSTM recurrence (WMMA bf16) --------
// One WG = 16 batch samples, 8 waves; wave w owns gate columns [64w, 64w+64).
// Combined weights live in registers (24 x v16bf per lane) for all T steps.
__global__ __launch_bounds__(256) void lstm_kernel(
    const float* __restrict__ x, const float* __restrict__ std_t,
    const unsigned short* __restrict__ wcomb, const float* __restrict__ biasg,
    const float* __restrict__ attnw, const float* __restrict__ attnb,
    unsigned short* __restrict__ hseq, float* __restrict__ logits) {
    __shared__ float gatesS[16 * G_];           // 32 KB
    __shared__ unsigned short hA[16 * KTOT];    // 6 KB  (A staging, bf16)
    __shared__ float biasS[G_];
    __shared__ float attnS[H_];
    __shared__ float part[16 * 16];

    const int tid  = threadIdx.x;
    const int lane = tid & 31;
    const int wave = tid >> 5;                  // 0..7
    const int b0   = blockIdx.x * 16;
    const int kgrp = lane >> 4;                 // 0/1 half-wave
    const int ln   = lane & 15;

    for (int i = tid; i < G_; i += 256) biasS[i] = biasg[i];
    for (int i = tid; i < H_; i += 256) attnS[i] = attnw[i];
    for (int i = tid; i < 16 * KTOT; i += 256) hA[i] = 0;   // h0 = 0, pad = 0
    __syncthreads();
    // x_0 + std_0 into A staging (bf16)
    {
        const int m = tid >> 4, c2 = (tid & 15) * 2;
        size_t xi = ((size_t)(b0 + m) * T_ + 0) * C_ + c2;
        unsigned int p = f2bfu(x[xi]) | (f2bfu(x[xi + 1]) << 16);
        ((unsigned int*)hA)[(m * KTOT + H_ + c2) >> 1] = p;
    }
    if (tid < 16) hA[tid * KTOT + H_ + C_] = (unsigned short)f2bfu(std_t[0]);

    // Load B fragments from global once: wcomb is [n][k] (k pairs contiguous).
    // B 32x16 bf16, this code's k order (same as A): lane col = ln, halves i -> K = kgrp*16 + i.
    v16bf Breg[24];
    const int ncb = wave * 64;
#pragma unroll
    for (int kc = 0; kc < 6; ++kc) {
#pragma unroll
        for (int nt = 0; nt < 4; ++nt) {
            const int n = ncb + nt * 16 + ln;
            U16 u;
#pragma unroll
            for (int v = 0; v < 8; ++v) {
                const int k = kc * 32 + kgrp * 16 + 2 * v;
                u.i[v] = *(const unsigned int*)(wcomb + (size_t)n * KTOT + k);
            }
            Breg[kc * 4 + nt] = u.v;
        }
    }
    __syncthreads();

    // elementwise mapping: thread -> (em, en0..en0+7) of h
    const int em  = tid & 15;
    const int en0 = (tid >> 4) * 8;
    float creg[8];
#pragma unroll
    for (int j = 0; j < 8; ++j) creg[j] = 0.f;
    const float attnbv = attnb[0];
    const unsigned int* hA32 = (const unsigned int*)hA;
    unsigned int* hA32w = (unsigned int*)hA;

    for (int t = 0; t < T_; ++t) {
        // ---- (a) gates tile = A(16x192) @ B(192x64-per-wave), f32 acc ----
        v8f acc[4];
        const v8f vz = {0.f, 0.f, 0.f, 0.f, 0.f, 0.f, 0.f, 0.f};
#pragma unroll
        for (int nt = 0; nt < 4; ++nt) acc[nt] = vz;
#pragma unroll
        for (int kc = 0; kc < 6; ++kc) {
            // A 16x32 bf16, this code's k order (same as B): row m = ln, halves i -> K = kgrp*16 + i
            U16 ua;
#pragma unroll
            for (int v = 0; v < 8; ++v) {
                const int i2 = 2 * v;
                const int k = kc * 32 + kgrp * 16 + i2;
                ua.i[v] = hA32[(ln * KTOT + k) >> 1];
            }
#pragma unroll
            for (int nt = 0; nt < 4; ++nt)
                acc[nt] = __builtin_amdgcn_wmma_f32_16x16x32_bf16(
                    false, ua.v, false, Breg[kc * 4 + nt],
                    (short)0, acc[nt], false, false);
        }
        // ---- (b) D tiles -> gates LDS (f32 C/D layout: vgpr r -> M=r+8*kgrp)
#pragma unroll
        for (int nt = 0; nt < 4; ++nt) {
            const int nc = ncb + nt * 16 + ln;
#pragma unroll
            for (int r = 0; r < 8; ++r)
                gatesS[(r + kgrp * 8) * G_ + nc] = acc[nt][r];
        }
        __syncthreads();
        // ---- (d) elementwise LSTM cell + h pack + logit partial ----
        float hv[8];
#pragma unroll
        for (int j = 0; j < 8; ++j) {
            const int n = en0 + j;
            const float gi = gatesS[em * G_ + n]            + biasS[n];
            const float gf = gatesS[em * G_ + H_ + n]       + biasS[H_ + n];
            const float gg = gatesS[em * G_ + 2 * H_ + n]   + biasS[2 * H_ + n];
            const float go = gatesS[em * G_ + 3 * H_ + n]   + biasS[3 * H_ + n];
            const float iv = 1.f / (1.f + expf(-gi));
            const float fv = 1.f / (1.f + expf(-gf));
            const float gv = tanhf(gg);
            const float ov = 1.f / (1.f + expf(-go));
            creg[j] = fv * creg[j] + iv * gv;
            hv[j]   = ov * tanhf(creg[j]);
        }
        unsigned int pk[4];
        float lp = 0.f;
#pragma unroll
        for (int j = 0; j < 4; ++j)
            pk[j] = f2bfu(hv[2 * j]) | (f2bfu(hv[2 * j + 1]) << 16);
#pragma unroll
        for (int j = 0; j < 8; ++j) lp += hv[j] * attnS[en0 + j];
        // h -> A staging for next step
#pragma unroll
        for (int j = 0; j < 4; ++j)
            hA32w[((em * KTOT + en0) >> 1) + j] = pk[j];
        // h -> global sequence (bf16, 16B per thread)
        {
            const size_t hi = ((size_t)(b0 + em) * T_ + t) * H_ + en0;
            uint4 v4; v4.x = pk[0]; v4.y = pk[1]; v4.z = pk[2]; v4.w = pk[3];
            *(uint4*)(hseq + hi) = v4;
        }
        part[em * 16 + (tid >> 4)] = lp;
        // prefetch x_{t+1} into A staging
        if (t + 1 < T_) {
            const int m = tid >> 4, c2 = (tid & 15) * 2;
            const size_t xi = ((size_t)(b0 + m) * T_ + (t + 1)) * C_ + c2;
            unsigned int p = f2bfu(x[xi]) | (f2bfu(x[xi + 1]) << 16);
            hA32w[(m * KTOT + H_ + c2) >> 1] = p;
            if (tid < 16) hA[tid * KTOT + H_ + C_] = (unsigned short)f2bfu(std_t[t + 1]);
        }
        __syncthreads();
        // ---- (f) attention logit reduce (16 lanes) ----
        if (tid < 16) {
            float s = 0.f;
#pragma unroll
            for (int j = 0; j < 16; ++j) s += part[tid * 16 + j];
            logits[(size_t)(b0 + tid) * T_ + t] = s + attnbv;
        }
    }
}

// ---------------- Kernel E: softmax-attention pooling + FC ----------------
__global__ __launch_bounds__(256) void pool_kernel(
    const float* __restrict__ logits, const unsigned short* __restrict__ hseq,
    const float* __restrict__ fcw, const float* __restrict__ fcb,
    float* __restrict__ out) {
    __shared__ float lg[T_];
    __shared__ float red[256];
    const int b = blockIdx.x, tid = threadIdx.x;
    lg[tid]       = logits[(size_t)b * T_ + tid];
    lg[tid + 256] = logits[(size_t)b * T_ + tid + 256];
    __syncthreads();
    red[tid] = fmaxf(lg[tid], lg[tid + 256]);
    __syncthreads();
    for (int s = 128; s > 0; s >>= 1) { if (tid < s) red[tid] = fmaxf(red[tid], red[tid + s]); __syncthreads(); }
    const float mx = red[0];
    __syncthreads();
    const float e0 = expf(lg[tid] - mx), e1 = expf(lg[tid + 256] - mx);
    lg[tid] = e0; lg[tid + 256] = e1;
    red[tid] = e0 + e1;
    __syncthreads();
    for (int s = 128; s > 0; s >>= 1) { if (tid < s) red[tid] += red[tid + s]; __syncthreads(); }
    const float inv = 1.f / red[0];
    lg[tid] *= inv; lg[tid + 256] *= inv;
    __syncthreads();
    // pooled[h] = sum_t a_t * h[b,t,h] ; two threads per h
    const int h = tid >> 1, half = tid & 1;
    float acc = 0.f;
    for (int t = half * 256; t < half * 256 + 256; ++t)
        acc += lg[t] * bf2f(hseq[((size_t)b * T_ + t) * H_ + h]);
    red[tid] = acc;
    __syncthreads();
    if (tid < H_) lg[tid] = (red[2 * tid] + red[2 * tid + 1]) * fcw[tid];
    __syncthreads();
    for (int s = 64; s > 0; s >>= 1) { if (tid < s) lg[tid] += lg[tid + s]; __syncthreads(); }
    if (tid == 0) out[b] = lg[0] + fcb[0];
}

// --------------------------------------------------------------------------
extern "C" void kernel_launch(void* const* d_in, const int* in_sizes, int n_in,
                              void* d_out, int out_size, void* d_ws, size_t ws_size,
                              hipStream_t stream) {
    const float* x      = (const float*)d_in[0];
    const float* w_ih   = (const float*)d_in[1];
    const float* u_ih   = (const float*)d_in[2];
    const float* w_hh   = (const float*)d_in[3];
    const float* u_hh   = (const float*)d_in[4];
    const float* b_ih   = (const float*)d_in[5];
    const float* b_hh   = (const float*)d_in[6];
    const float* attn_w = (const float*)d_in[7];
    const float* attn_b = (const float*)d_in[8];
    const float* fc_w   = (const float*)d_in[9];
    const float* u_fc   = (const float*)d_in[10];
    const float* fc_b   = (const float*)d_in[11];
    float* out = (float*)d_out;

    char* ws = (char*)d_ws;
    float* std_t          = (float*)(ws + 0);                  // 2048 B
    float* sig            = (float*)(ws + 2048);               // 16 B
    float* biasg          = (float*)(ws + 2112);               // 2048 B
    float* fcw            = (float*)(ws + 4160);               // 512 B
    unsigned short* wcomb = (unsigned short*)(ws + 4672);      // 196608 B
    float* logits         = (float*)(ws + 201280);             // 1 MB
    unsigned short* hseq  = (unsigned short*)(ws + 201280 + 1048576); // 64 MB

    std_kernel<<<T_, 256, 0, stream>>>(x, std_t);
    sn_kernel<<<3, 256, 0, stream>>>(w_ih, u_ih, w_hh, u_hh, fc_w, u_fc, sig);
    pack_kernel<<<(G_ * KTOT) / 256, 256, 0, stream>>>(w_ih, w_hh, b_ih, b_hh,
                                                       fc_w, sig, wcomb, biasg, fcw);
    lstm_kernel<<<B_ / 16, 256, 0, stream>>>(x, std_t, wcomb, biasg,
                                             attn_w, attn_b, hseq, logits);
    pool_kernel<<<B_, 256, 0, stream>>>(logits, hseq, fcw, fc_b, out);
}
